// MyNeuralNetwork_62165356642731
// MI455X (gfx1250) — compile-verified, matrix operand order checked by reading, not run on hardware
//
#include <hip/hip_runtime.h>

typedef __attribute__((ext_vector_type(16))) _Float16 v16h;
typedef __attribute__((ext_vector_type(8)))  _Float16 v8h;
typedef __attribute__((ext_vector_type(8)))  float    v8f;

#define BATCH   2048
#define NOUT    15
#define NPATCH  225
#define PDIM    27

// K-index of element e in a 16-bit WMMA A/B fragment (16x32 / 32x16):
// hi=0 lanes: V0..V3 -> K=0..7, V4..V7 -> K=16..23
// hi=1 lanes: V0..V3 -> K=8..15, V4..V7 -> K=24..31
__device__ __forceinline__ int kmap(int e, int hi) {
    return ((e < 8) ? e : e + 8) + hi * 8;
}

// ---------------------------------------------------------------------------
// Weight prep: f32 W[N][K] -> f16 Wh[N][Kp], zero padded (Kp = roundup32(K))
// ---------------------------------------------------------------------------
__global__ __launch_bounds__(256)
void wprep_kernel(const float* __restrict__ W, _Float16* __restrict__ Wh,
                  int K, int Kp, int total) {
    int i = blockIdx.x * 256 + threadIdx.x;
    if (i >= total) return;
    int n = i / Kp, k = i - n * Kp;
    Wh[i] = (k < K) ? (_Float16)W[n * K + k] : (_Float16)0.0f;
}

// ---------------------------------------------------------------------------
// Kernel 1: SOM distance GEMM (WMMA M=225/img, N=256, K=27 pad 32) + argmin.
// One workgroup (8 waves) per image; all loads unconditional (clamped).
// ---------------------------------------------------------------------------
__global__ __launch_bounds__(256)
void som_wmma_kernel(const float* __restrict__ x,     // [B,3,32,32]
                     const float* __restrict__ som,   // [256,27]
                     _Float16*    __restrict__ h0)    // [B,480] f16 (pad zeroed)
{
    const int b    = blockIdx.x;
    const int lane = threadIdx.x & 31;
    const int wave = threadIdx.x >> 5;
    const int lid  = lane & 15;
    const int hi   = lane >> 4;

    // ---- Preload codebook B-fragments; derive ||s||^2 via cross-half shuffle
    v16h  Breg[16];
    float s2[16];
    #pragma unroll
    for (int t = 0; t < 16; ++t) {
        const float* sp = som + (t * 16 + lid) * PDIM;
        v16h bf;
        float part = 0.0f;
        #pragma unroll
        for (int e = 0; e < 16; ++e) {
            const int k  = kmap(e, hi);
            const int kc = (k < PDIM) ? k : 0;          // clamp -> valid address
            const float vf = sp[kc];                    // unconditional load
            const _Float16 v = (k < PDIM) ? (_Float16)vf : (_Float16)0.0f;
            bf[e] = v;
            const float q = (float)v;
            part += q * q;
        }
        Breg[t] = bf;
        // lanes (lid) and (lid+16) jointly hold all 27 K's of the same cell
        s2[t] = part + __shfl_xor(part, 16, 32);
    }

    const float* xb = x + (size_t)b * 3 * 32 * 32;

    for (int it = 0; it < 2; ++it) {
        const int mt = wave + 8 * it;
        if (mt >= 15) break;                            // wave-uniform

        const int p  = mt * 16 + lid;
        const bool valid_row = (p < NPATCH);
        const int pc = valid_row ? p : 0;
        const int oy = pc / NOUT, ox = pc - (pc / NOUT) * NOUT;
        const int base = oy * 64 + ox * 2;              // (2oy)*32 + 2ox

        v16h a;
        #pragma unroll
        for (int e = 0; e < 16; ++e) {
            const int k  = kmap(e, hi);
            const int kc = (k < PDIM) ? k : 0;
            const int c  = kc / 9, rem = kc - c * 9;
            const int kh = rem / 3, kw = rem - (rem / 3) * 3;
            const float vf = xb[base + c * 1024 + kh * 32 + kw];  // unconditional
            a[e] = (k < PDIM && valid_row) ? (_Float16)vf : (_Float16)0.0f;
        }

        float minv[8];
        int   mini[8];
        #pragma unroll
        for (int r = 0; r < 8; ++r) { minv[r] = 3.0e38f; mini[r] = 0; }

        #pragma unroll
        for (int t = 0; t < 16; ++t) {
            v8f c = {};
            c = __builtin_amdgcn_wmma_f32_16x16x32_f16(
                    false, a, false, Breg[t], (short)0, c, false, false);
            const int cell = t * 16 + lid;
            const float sv = s2[t];
            #pragma unroll
            for (int r = 0; r < 8; ++r) {
                const float d = fmaf(-2.0f, c[r], sv);
                if (d < minv[r]) { minv[r] = d; mini[r] = cell; }
            }
        }

        // argmin across the 16-lane half (rows 0-7 in hi=0, rows 8-15 in hi=1)
        #pragma unroll
        for (int m = 1; m < 16; m <<= 1) {
            #pragma unroll
            for (int r = 0; r < 8; ++r) {
                const float ov = __shfl_xor(minv[r], m, 16);
                const int   oi = __shfl_xor(mini[r], m, 16);
                if (ov < minv[r]) { minv[r] = ov; mini[r] = oi; }
            }
        }

        #pragma unroll
        for (int r = 0; r < 8; ++r) {
            if (lid == r) {
                const int pw = mt * 16 + r + 8 * hi;
                if (pw < NPATCH) {
                    const int win = mini[r];
                    _Float16* hb = h0 + (size_t)b * 480;
                    hb[pw]          = (_Float16)((float)(win >> 4) * 0.0625f);
                    hb[NPATCH + pw] = (_Float16)((float)(win & 15) * 0.0625f);
                }
            }
        }
    }
}

// ---------------------------------------------------------------------------
// Kernel 2: WMMA GEMM  out = [relu](A[M,Kp](f16) * Wh[N,Kp]^T + b)
// Fragments load as 2x b128 per matrix per K-step (layout-native).
// ---------------------------------------------------------------------------
__global__ __launch_bounds__(256)
void mlp_wmma_kernel(const _Float16* __restrict__ A,    // [M][Kp] f16, padded
                     const _Float16* __restrict__ Wh,   // [N][Kp] f16, padded
                     const float*    __restrict__ bias, // [N]
                     _Float16*       __restrict__ out16,// [M][Np] f16 (or null)
                     float*          __restrict__ out32,// [M][N]  f32 (or null)
                     int Kp, int N, int Np, int doRelu)
{
    const int lane = threadIdx.x & 31;
    const int wave = threadIdx.x >> 5;
    const int lid  = lane & 15;
    const int hi   = lane >> 4;
    const int row0 = (blockIdx.x * 8 + wave) * 16;

    const _Float16* Ab = A + (size_t)(row0 + lid) * Kp + hi * 8;
    const int nT = (N + 15) >> 4;
    const int kS = Kp >> 5;

    for (int nt = 0; nt < nT; ++nt) {
        const int ncol = nt * 16 + lid;
        const int nc   = (ncol < N) ? ncol : (N - 1);
        const _Float16* Wb = Wh + (size_t)nc * Kp + hi * 8;

        v8f acc = {};
        for (int ks = 0; ks < kS; ++ks) {
            const int k0 = ks * 32;
            const v8h a0 = *(const v8h*)(Ab + k0);
            const v8h a1 = *(const v8h*)(Ab + k0 + 16);
            const v8h b0 = *(const v8h*)(Wb + k0);
            const v8h b1 = *(const v8h*)(Wb + k0 + 16);
            const v16h af = __builtin_shufflevector(a0, a1,
                0,1,2,3,4,5,6,7,8,9,10,11,12,13,14,15);
            const v16h bf = __builtin_shufflevector(b0, b1,
                0,1,2,3,4,5,6,7,8,9,10,11,12,13,14,15);
            acc = __builtin_amdgcn_wmma_f32_16x16x32_f16(
                      false, af, false, bf, (short)0, acc, false, false);
        }

        if (ncol < N) {
            const float bv = bias[ncol];
            #pragma unroll
            for (int r = 0; r < 8; ++r) {
                float v = acc[r] + bv;
                if (doRelu) v = (v > 0.0f) ? v : 0.0f;
                const int row = row0 + r + 8 * hi;
                if (out16) out16[(size_t)row * Np + ncol] = (_Float16)v;
                else       out32[(size_t)row * N  + ncol] = v;
            }
        }
    }
}

// ---------------------------------------------------------------------------
// Host launcher
// ---------------------------------------------------------------------------
extern "C" void kernel_launch(void* const* d_in, const int* in_sizes, int n_in,
                              void* d_out, int out_size, void* d_ws, size_t ws_size,
                              hipStream_t stream) {
    const float* x   = (const float*)d_in[0];
    const float* som = (const float*)d_in[1];
    const float* W1  = (const float*)d_in[2];
    const float* b1  = (const float*)d_in[3];
    const float* W2  = (const float*)d_in[4];
    const float* b2  = (const float*)d_in[5];
    const float* W3  = (const float*)d_in[6];
    const float* b3  = (const float*)d_in[7];
    const float* W4  = (const float*)d_in[8];
    const float* b4  = (const float*)d_in[9];
    float* out = (float*)d_out;

    // Padded K per layer (roundup to 32)
    //   h0: [2048][480]  h1: [2048][416]  h2: [2048][224]  h3: [2048][128]
    char* ws = (char*)d_ws;
    _Float16* h0  = (_Float16*)(ws + 0);
    _Float16* h1  = (_Float16*)(ws + 1966080);
    _Float16* h2  = (_Float16*)(ws + 3670016);
    _Float16* h3  = (_Float16*)(ws + 4587520);
    const size_t actBytes = 5111808;                    // h0..h3 total
    _Float16* Wh1 = (_Float16*)(ws + 5111808);          // [400][480]
    _Float16* Wh2 = (_Float16*)(ws + 5495808);          // [200][416]
    _Float16* Wh3 = (_Float16*)(ws + 5662208);          // [100][224]
    _Float16* Wh4 = (_Float16*)(ws + 5707008);          // [ 10][128]

    // Zero activation buffers (covers all K-padding) — graph-capturable
    hipMemsetAsync(ws, 0, actBytes, stream);

    // Weight conversion/pad to f16
    wprep_kernel<<<(400*480 + 255)/256, 256, 0, stream>>>(W1, Wh1, 450, 480, 400*480);
    wprep_kernel<<<(200*416 + 255)/256, 256, 0, stream>>>(W2, Wh2, 400, 416, 200*416);
    wprep_kernel<<<(100*224 + 255)/256, 256, 0, stream>>>(W3, Wh3, 200, 224, 100*224);
    wprep_kernel<<<( 10*128 + 255)/256, 256, 0, stream>>>(W4, Wh4, 100, 128,  10*128);

    // SOM feature stage: one workgroup per image
    som_wmma_kernel<<<BATCH, 256, 0, stream>>>(x, som, h0);

    // MLP stage: 128 row-tiles -> 16 workgroups of 8 waves
    mlp_wmma_kernel<<<16, 256, 0, stream>>>(h0, Wh1, b1, h1, nullptr, 480, 400, 416, 1);
    mlp_wmma_kernel<<<16, 256, 0, stream>>>(h1, Wh2, b2, h2, nullptr, 416, 200, 224, 1);
    mlp_wmma_kernel<<<16, 256, 0, stream>>>(h2, Wh3, b3, h3, nullptr, 224, 100, 128, 1);
    mlp_wmma_kernel<<<16, 256, 0, stream>>>(h3, Wh4, b4, nullptr, out,  128,  10,  10, 0);
}
